// AugmentedLayer_18184891532168
// MI455X (gfx1250) — compile-verified, matrix operand order checked by reading, not run on hardware
//
#include <hip/hip_runtime.h>

typedef __attribute__((ext_vector_type(16))) _Float16 v16h;
typedef __attribute__((ext_vector_type(8)))  _Float16 v8h;
typedef __attribute__((ext_vector_type(8)))  float    v8f;

#define QSC 0.35355339059327373f   // 8^-0.5

static __device__ __forceinline__ v16h pack16(v8h lo, v8h hi) {
  v16h r;
#pragma unroll
  for (int j = 0; j < 8; ++j) { r[j] = lo[j]; r[j + 8] = hi[j]; }
  return r;
}

// ---------------------------------------------------------------------------
// Prep A: x [b][64c][1024p] f32 -> xT [b][1024p][64c] f16
// ---------------------------------------------------------------------------
__global__ void xt_prep_kernel(const float* __restrict__ x,
                               _Float16* __restrict__ xT) {
  int idx = blockIdx.x * 256 + threadIdx.x;   // 8*64*1024
  int b = idx >> 16;
  int r = idx & 65535;
  int c = r >> 10;
  int p = r & 1023;
  xT[(((b << 10) + p) << 6) + c] = (_Float16)x[idx];
}

// ---------------------------------------------------------------------------
// Prep B: conv_w [oc][c][3][3] f32 -> wT [tap][oc][c] f16
// ---------------------------------------------------------------------------
__global__ void wt_prep_kernel(const float* __restrict__ conv_w,
                               _Float16* __restrict__ wT) {
  int idx = blockIdx.x * 256 + threadIdx.x;   // 9*64*64 = 36864
  int tap = idx >> 12;
  int r   = idx & 4095;
  int oc  = r >> 6, c = r & 63;
  wT[idx] = (_Float16)conv_w[(oc * 64 + c) * 9 + tap];
}

// ---------------------------------------------------------------------------
// QKV 1x1 as WMMA GEMM: [192oc x 64c] x [64c x 8192px]
// ---------------------------------------------------------------------------
__global__ __launch_bounds__(256, 1)
void qkv_gemm_kernel(const _Float16* __restrict__ xT,
                     const float* __restrict__ qkv_w,
                     const float* __restrict__ qkv_b,
                     _Float16* __restrict__ QKVt) {
  int gid  = blockIdx.x * 8 + (threadIdx.x >> 5);   // 8*12*64 = 6144 waves
  int lane = threadIdx.x & 31;
  int hl = lane >> 4, n = lane & 15, mA = lane & 15;
  int b  = gid / (12 * 64);
  int r2 = gid - b * (12 * 64);
  int octile = r2 >> 6, pt = r2 & 63;
  int oc0   = octile << 4;
  int pixel = (pt << 4) + n;

  const float* wr = qkv_w + (oc0 + mA) * 64;
  const v8h*   xr = (const v8h*)(xT + (((b << 10) + pixel) << 6));

  v8f acc = {0.f, 0.f, 0.f, 0.f, 0.f, 0.f, 0.f, 0.f};
#pragma unroll
  for (int kh = 0; kh < 2; ++kh) {
    v8h alo, ahi;
#pragma unroll
    for (int j = 0; j < 8; ++j) {
      alo[j] = (_Float16)wr[(hl << 3) + (kh << 5) + j];
      ahi[j] = (_Float16)wr[(hl << 3) + (kh << 5) + 16 + j];
    }
    v16h A = pack16(alo, ahi);
    v16h B = pack16(xr[hl + (kh << 2)], xr[hl + (kh << 2) + 2]);
    acc = __builtin_amdgcn_wmma_f32_16x16x32_f16(false, A, false, B,
                                                 (short)0, acc, false, false);
  }
  float scale = (octile < 4) ? QSC : 1.0f;
#pragma unroll
  for (int r = 0; r < 8; ++r) {
    int o = oc0 + r + (hl << 3);
    float v = (acc[r] + qkv_b[o]) * scale;
    QKVt[(b * 192 + o) * 1024 + pixel] = (_Float16)v;
  }
}

// ---------------------------------------------------------------------------
// Conv 3x3 as implicit WMMA GEMM: 9 taps x (K=64) = 18 WMMAs per 16x16 tile.
// ---------------------------------------------------------------------------
__global__ __launch_bounds__(256, 1)
void conv_gemm_kernel(const _Float16* __restrict__ xT,
                      const _Float16* __restrict__ wT,
                      const float* __restrict__ conv_b,
                      float* __restrict__ out) {
  __shared__ __align__(16) _Float16 xs[6 * 34 * 64];   // 25.5 KB
  __shared__ __align__(16) _Float16 wsh[9 * 16 * 64];  // 18 KB

  int tid = threadIdx.x;
  int wv = tid >> 5, lane = tid & 31;
  int hl = lane >> 4, n = lane & 15, mA = lane & 15;
  int blk = blockIdx.x;               // 8b * 4octile * 8pgroup = 256
  int b  = blk >> 5;
  int r2 = blk & 31;
  int octile = r2 >> 3, pb = r2 & 7;
  int oc0 = octile << 4;
  int h0  = pb << 2;

  for (int i = tid; i < 6 * 34 * 64; i += 256) {
    int row = i / (34 * 64);
    int rr  = i - row * (34 * 64);
    int col = rr >> 6, c = rr & 63;
    int hh = h0 - 1 + row, ww = col - 1;
    bool ok = (hh >= 0) && (hh < 32) && (ww >= 0) && (ww < 32);
    xs[i] = ok ? xT[(((b << 10) + (hh << 5) + ww) << 6) + c] : (_Float16)0.f;
  }
  for (int i = tid; i < 9 * 16 * 64; i += 256) {
    int tap = i >> 10;
    int rr  = i & 1023;
    int ocl = rr >> 6, c = rr & 63;
    wsh[i] = wT[(((tap << 6) + oc0 + ocl) << 6) + c];
  }
  __syncthreads();

  int rowl = wv >> 1;
  int w0   = (wv & 1) << 4;
  v8f acc = {0.f, 0.f, 0.f, 0.f, 0.f, 0.f, 0.f, 0.f};
#pragma unroll
  for (int kh = 0; kh < 3; ++kh) {
#pragma unroll
    for (int kw = 0; kw < 3; ++kw) {
      int tap = kh * 3 + kw;
      const v8h* wrow = (const v8h*)&wsh[((tap << 4) + mA) << 6];
      const v8h* xrow = (const v8h*)&xs[(((rowl + kh) * 34) + (w0 + n + kw)) << 6];
#pragma unroll
      for (int k2 = 0; k2 < 2; ++k2) {
        v16h A = pack16(wrow[hl + (k2 << 2)], wrow[hl + (k2 << 2) + 2]);
        v16h B = pack16(xrow[hl + (k2 << 2)], xrow[hl + (k2 << 2) + 2]);
        acc = __builtin_amdgcn_wmma_f32_16x16x32_f16(false, A, false, B,
                                                     (short)0, acc, false, false);
      }
    }
  }
  int p = ((h0 + rowl) << 5) + w0 + n;
#pragma unroll
  for (int r = 0; r < 8; ++r) {
    int oc = oc0 + r + (hl << 3);
    out[b * 131072 + oc * 1024 + p] = acc[r] + conv_b[oc];
  }
}

// ---------------------------------------------------------------------------
// Fused attention with relative position logits.
//   V staged via async global->LDS copy (ASYNCcnt); rest synchronous.
// ---------------------------------------------------------------------------
__global__ __launch_bounds__(256, 1)
void attn_kernel(const _Float16* __restrict__ QKVt,
                 const float* __restrict__ key_rel_w,
                 const float* __restrict__ key_rel_h,
                 _Float16* __restrict__ A_pre) {
  __shared__ __align__(16) _Float16 KM[(1024 + 64) * 8];   // 17 KB
  __shared__ __align__(16) _Float16 Vt[8][1024];           // 16 KB
  __shared__ __align__(16) _Float16 Qs[128 * 8];           //  2 KB
  __shared__ __align__(16) _Float16 krw[64 * 8];           //  1 KB
  __shared__ __align__(16) _Float16 Ps[8][16][32];         //  8 KB

  const int bh   = blockIdx.x >> 3;
  const int part = blockIdx.x & 7;
  const int tid  = threadIdx.x;
  const int wv   = tid >> 5;
  const int lane = tid & 31;
  const int hl   = lane >> 4;
  const int n    = lane & 15;
  const int b    = bh >> 3, nh = bh & 7;
  const int nh8  = nh << 3;

  {
    const _Float16* base = QKVt + b * 192 * 1024;
    // V: source rows (128+nh8+d) are contiguous -> straight 16 KB linear copy.
    // Use async global->LDS (bypasses VGPRs, tracked by ASYNCcnt).
    {
      const char* Vg  = (const char*)(base + (128 + nh8) * 1024);
      unsigned    vtb = (unsigned)(uintptr_t)(&Vt[0][0]);
      for (int i = tid; i < 1024; i += 256) {          // 1024 x 16 B
        unsigned    lds_a = vtb + i * 16;
        const void* g     = Vg + i * 16;
        asm volatile("global_load_async_to_lds_b128 %0, %1, off"
                     :: "v"(lds_a), "v"(g) : "memory");
      }
    }
    for (int i = tid; i < 8192; i += 256) {            // K: d-outer, coalesced
      int d = i >> 10, key = i & 1023;
      KM[key * 8 + d] = base[(64 + nh8 + d) * 1024 + key];
    }
    for (int i = tid; i < 1024; i += 256) {
      int d = i >> 7, ql = i & 127;
      Qs[ql * 8 + d] = base[(nh8 + d) * 1024 + part * 128 + ql];
    }
    for (int i = tid; i < 63 * 8; i += 256) {
      krw[i] = (_Float16)key_rel_w[i];
      KM[1024 * 8 + i] = (_Float16)key_rel_h[i];
    }
  }
  asm volatile("s_wait_asynccnt 0x0" ::: "memory");
  __syncthreads();

  const int qg0 = (part << 7) + (wv << 4);
  const int hq  = qg0 >> 5;
  const int w0  = qg0 & 31;
  const int ql0 = wv << 4;
  const int mA  = lane & 15;

  v16h aq;
  {
    v8h qrow = *(const v8h*)&Qs[(ql0 + mA) * 8];
#pragma unroll
    for (int j = 0; j < 8; ++j) aq[j] = qrow[j];
#pragma unroll
    for (int j = 8; j < 16; ++j) aq[j] = (_Float16)0.0f;
  }

  float rwv[2][8];
#pragma unroll
  for (int r = 0; r < 8; ++r) {
    int m = r + (hl << 3);
    v8h qrow = *(const v8h*)&Qs[(ql0 + m) * 8];
#pragma unroll
    for (int s = 0; s < 2; ++s) {
      int ridx = (s * 16 + n) - (w0 + m) + 31;   // always in [0,62]
      v8h wrow = *(const v8h*)&krw[ridx * 8];
      float acc = 0.f;
#pragma unroll
      for (int j = 0; j < 8; ++j) acc += (float)qrow[j] * (float)wrow[j];
      rwv[s][r] = acc;
    }
  }

  v8f o_acc = {0.f, 0.f, 0.f, 0.f, 0.f, 0.f, 0.f, 0.f};

  for (int it = 0; it < 32; ++it) {
    const int k0   = it << 5;
    const int hidx = it - hq + 31;
    const int rowH = 1024 + hidx;
    v8h kv0 = *(const v8h*)&KM[(hl ? rowH : (k0 + n)) * 8];
    v8h kv1 = *(const v8h*)&KM[(hl ? rowH : (k0 + 16 + n)) * 8];
    v16h b0, b1;
#pragma unroll
    for (int j = 0; j < 8; ++j) { b0[j] = kv0[j]; b1[j] = kv1[j]; }
#pragma unroll
    for (int j = 8; j < 16; ++j) { b0[j] = (_Float16)0.f; b1[j] = (_Float16)0.f; }

    v8f cz = {0.f, 0.f, 0.f, 0.f, 0.f, 0.f, 0.f, 0.f};
    v8f c0 = __builtin_amdgcn_wmma_f32_16x16x32_f16(false, aq, false, b0,
                                                    (short)0, cz, false, false);
    v8f c1 = __builtin_amdgcn_wmma_f32_16x16x32_f16(false, aq, false, b1,
                                                    (short)0, cz, false, false);
#pragma unroll
    for (int r = 0; r < 8; ++r) {
      c0[r] = __expf(c0[r] + rwv[0][r]);
      c1[r] = __expf(c1[r] + rwv[1][r]);
    }
#pragma unroll
    for (int r = 0; r < 8; ++r) {
      int m = r + (hl << 3);
      Ps[wv][m][n]      = (_Float16)c0[r];
      Ps[wv][m][16 + n] = (_Float16)c1[r];
    }
    v8h alo = *(const v8h*)&Ps[wv][mA][hl << 3];
    v8h ahi = *(const v8h*)&Ps[wv][mA][16 + (hl << 3)];
    v16h ap = pack16(alo, ahi);

    v8h vlo = *(const v8h*)&Vt[n & 7][k0 + (hl << 3)];
    v8h vhi = *(const v8h*)&Vt[n & 7][k0 + 16 + (hl << 3)];
    _Float16 fill = (n == 8) ? (_Float16)1.0f : (_Float16)0.0f;
    v16h bv;
#pragma unroll
    for (int j = 0; j < 8; ++j) {
      bv[j]     = (n < 8) ? vlo[j] : fill;
      bv[j + 8] = (n < 8) ? vhi[j] : fill;
    }
    o_acc = __builtin_amdgcn_wmma_f32_16x16x32_f16(false, ap, false, bv,
                                                   (short)0, o_acc, false, false);
  }

#pragma unroll
  for (int r = 0; r < 8; ++r) {
    float rs  = __shfl(o_acc[r], (hl << 4) + 8, 32);
    float val = o_acc[r] / rs;
    if (n < 8) {
      int m = r + (hl << 3);
      A_pre[(((b << 10) + qg0 + m) << 6) + nh8 + n] = (_Float16)val;
    }
  }
}

// ---------------------------------------------------------------------------
// Final 1x1 projection as WMMA GEMM: [64oc x 64c] x [64c x 8192px]
// ---------------------------------------------------------------------------
__global__ __launch_bounds__(256, 1)
void attn_proj_gemm_kernel(const _Float16* __restrict__ Ap,
                           const float* __restrict__ attn_w,
                           const float* __restrict__ attn_b,
                           float* __restrict__ out) {
  int gid  = blockIdx.x * 8 + (threadIdx.x >> 5);   // 8*4*64 = 2048 waves
  int lane = threadIdx.x & 31;
  int hl = lane >> 4, n = lane & 15, mA = lane & 15;
  int b  = gid >> 8;
  int r2 = gid & 255;
  int octile = r2 >> 6, pt = r2 & 63;
  int oc0   = octile << 4;
  int pixel = (pt << 4) + n;

  const float* wr = attn_w + (oc0 + mA) * 64;
  const v8h*   xr = (const v8h*)(Ap + (((b << 10) + pixel) << 6));

  v8f acc = {0.f, 0.f, 0.f, 0.f, 0.f, 0.f, 0.f, 0.f};
#pragma unroll
  for (int kh = 0; kh < 2; ++kh) {
    v8h alo, ahi;
#pragma unroll
    for (int j = 0; j < 8; ++j) {
      alo[j] = (_Float16)wr[(hl << 3) + (kh << 5) + j];
      ahi[j] = (_Float16)wr[(hl << 3) + (kh << 5) + 16 + j];
    }
    v16h A = pack16(alo, ahi);
    v16h B = pack16(xr[hl + (kh << 2)], xr[hl + (kh << 2) + 2]);
    acc = __builtin_amdgcn_wmma_f32_16x16x32_f16(false, A, false, B,
                                                 (short)0, acc, false, false);
  }
#pragma unroll
  for (int r = 0; r < 8; ++r) {
    int oc = oc0 + r + (hl << 3);
    out[b * 131072 + (64 + oc) * 1024 + pixel] = acc[r] + attn_b[oc];
  }
}

// ---------------------------------------------------------------------------
extern "C" void kernel_launch(void* const* d_in, const int* in_sizes, int n_in,
                              void* d_out, int out_size, void* d_ws, size_t ws_size,
                              hipStream_t stream) {
  (void)in_sizes; (void)n_in; (void)out_size; (void)ws_size;
  const float* x      = (const float*)d_in[0];
  const float* conv_w = (const float*)d_in[1];
  const float* conv_b = (const float*)d_in[2];
  const float* qkv_w  = (const float*)d_in[3];
  const float* qkv_b  = (const float*)d_in[4];
  const float* attn_w = (const float*)d_in[5];
  const float* attn_b = (const float*)d_in[6];
  const float* krw    = (const float*)d_in[7];
  const float* krh    = (const float*)d_in[8];
  float* out = (float*)d_out;

  char* ws = (char*)d_ws;
  _Float16* xT   = (_Float16*)(ws);                  // 1 MiB : [b][p][64] f16
  _Float16* wT   = (_Float16*)(ws + 0x100000);       // 72 KiB: [tap][oc][c] f16
  _Float16* QKVt = (_Float16*)(ws + 0x120000);       // 3 MiB : [b][192][1024] f16
  _Float16* Apre = (_Float16*)(ws + 0x420000);       // 1 MiB : [b][p][64] f16

  xt_prep_kernel<<<2048, 256, 0, stream>>>(x, xT);
  wt_prep_kernel<<<144, 256, 0, stream>>>(conv_w, wT);
  qkv_gemm_kernel<<<768, 256, 0, stream>>>(xT, qkv_w, qkv_b, QKVt);
  conv_gemm_kernel<<<256, 256, 0, stream>>>(xT, wT, conv_b, out);
  attn_kernel<<<512, 256, 0, stream>>>(QKVt, krw, krh, Apre);
  attn_proj_gemm_kernel<<<256, 256, 0, stream>>>(Apre, attn_w, attn_b, out);
}
